// LSTMClassifier_9302899163248
// MI455X (gfx1250) — compile-verified, matrix operand order checked by reading, not run on hardware
//
#include <hip/hip_runtime.h>

typedef __attribute__((ext_vector_type(16))) _Float16 v16h;
typedef __attribute__((ext_vector_type(8)))  _Float16 v8h;
typedef __attribute__((ext_vector_type(8)))  float    v8f;

#define BSZ   2048
#define TLEN  256
#define FDIM  64
#define HDIM  128
#define G4H   512   // 4*H
#define HSTR  136   // padded h row stride (halves): multiple of 8, not of 128

// ---------------------------------------------------------------------------
// DPP quad_perm lane exchange (single VALU op, no LDS).
// 0xB1 = [1,0,3,2] (xor 1), 0x4E = [2,3,0,1] (xor 2)
// ---------------------------------------------------------------------------
template <int CTRL>
__device__ __forceinline__ float qperm(float v) {
  return __builtin_bit_cast(float,
      __builtin_amdgcn_update_dpp(0, __builtin_bit_cast(int, v),
                                  CTRL, 0xF, 0xF, true));
}

__device__ __forceinline__ float fast_sigmoid(float x) {
  return __builtin_amdgcn_rcpf(1.f + __expf(-x));
}
__device__ __forceinline__ float fast_tanh(float x) {
  return fmaf(2.f, __builtin_amdgcn_rcpf(1.f + __expf(-2.f * x)), -1.f);
}

// ---------------------------------------------------------------------------
// Prep: convert weights to f16 with gate-interleaved row permutation
//   n_new = j*4 + gate   (gate: 0=i,1=f,2=g,3=o; original row = gate*128 + j)
// and fold biases (same permutation).
// ---------------------------------------------------------------------------
__global__ void lstm_prep_kernel(const float* __restrict__ W_ih,
                                 const float* __restrict__ W_hh,
                                 const float* __restrict__ b_ih,
                                 const float* __restrict__ b_hh,
                                 _Float16* __restrict__ Wih_h,
                                 _Float16* __restrict__ Whh_h,
                                 float* __restrict__ bias) {
  int i = blockIdx.x * blockDim.x + threadIdx.x;
  const int NIH = G4H * FDIM;          // 32768
  const int NHH = G4H * HDIM;          // 65536
  if (i < NIH) {
    int n = i >> 6, k = i & 63;
    int j = n >> 2, G = n & 3;
    Wih_h[i] = (_Float16)W_ih[(G * HDIM + j) * FDIM + k];
  } else if (i < NIH + NHH) {
    int v = i - NIH;
    int n = v >> 7, k = v & 127;
    int j = n >> 2, G = n & 3;
    Whh_h[v] = (_Float16)W_hh[(G * HDIM + j) * HDIM + k];
  } else if (i < NIH + NHH + G4H) {
    int n = i - NIH - NHH;
    int j = n >> 2, G = n & 3;
    bias[n] = b_ih[G * HDIM + j] + b_hh[G * HDIM + j];
  }
}

// ---------------------------------------------------------------------------
// A-fragment loader: 16x32 f16 tile, one WMMA operand per lane.
// lane 0..15  : row m=lane,    K = kbase + {0..7, 16..23}
// lane 16..31 : row m=lane-16, K = kbase + {8..15, 24..31}  (kbase includes +8)
// ---------------------------------------------------------------------------
__device__ __forceinline__ v16h load_a_frag(const _Float16* base, int stride,
                                            int m, int kbase) {
  const _Float16* p = base + m * stride + kbase;
  v8h lo = *(const v8h*)(p);
  v8h hi = *(const v8h*)(p + 16);
  v16h r;
#pragma unroll
  for (int i = 0; i < 8; ++i) { r[i] = lo[i]; r[i + 8] = hi[i]; }
  return r;
}

// ---------------------------------------------------------------------------
// Main LSTM kernel: one block = 16 batch rows, 512 threads = 16 waves.
// Gate-interleaved weights -> each wave owns 8 full hidden units; the cell
// update is wave-private (DPP quad transpose), c lives in registers,
// h/x are double-buffered in LDS -> ONE barrier per timestep.
// ---------------------------------------------------------------------------
__launch_bounds__(512)
__global__ void lstm_main_kernel(const float* __restrict__ x,
                                 const _Float16* __restrict__ Wih_h,
                                 const _Float16* __restrict__ Whh_h,
                                 const float* __restrict__ bias,
                                 float* __restrict__ h_last) {
  __shared__ __align__(16) _Float16 x_lds[2][16 * FDIM];   // x tile, double buf
  __shared__ __align__(16) _Float16 h_lds[2][16 * HSTR];   // h state, double buf

  const int tid   = threadIdx.x;
  const int lane  = tid & 31;
  const int wave  = tid >> 5;            // 0..15
  const int b0    = blockIdx.x * 16;     // batch tile start
  const int lm    = lane & 15;           // A row / B column within tile
  const int hi    = lane >> 4;           // lane half
  const int khalf = hi * 8;              // lane-group K shift
  const int gl    = lane & 3;            // gate index this lane holds pre-transpose
  const int jj    = lm >> 2;             // hidden-unit sub-index within tile
  const int mrow  = 8 * hi + 2 * gl;     // rows handled post-transpose: mrow, mrow+1

  // zero-init h buffer 0 (read at t=0)
  for (int u = tid; u < 16 * HSTR; u += 512) h_lds[0][u] = (_Float16)0.f;

  // per-lane bias for this wave's two gate columns (C layout: whole lane = one n)
  float bias_n[2];
#pragma unroll
  for (int nt = 0; nt < 2; ++nt) bias_n[nt] = bias[wave * 32 + nt * 16 + lm];

  // ---- Preload this wave's B fragments (permuted weights) into registers ----
  v16h bih[2][2];   // x-part:  2 N-tiles x 2 K-tiles (F=64)
  v16h bhh[2][4];   // h-part:  2 N-tiles x 4 K-tiles (H=128)
#pragma unroll
  for (int nt = 0; nt < 2; ++nt) {
    const int n = wave * 32 + nt * 16 + lm;   // permuted gate column
#pragma unroll
    for (int kt = 0; kt < 2; ++kt) {
      const _Float16* p = Wih_h + n * FDIM + kt * 32 + khalf;
      v8h lo = *(const v8h*)(p);
      v8h hh = *(const v8h*)(p + 16);
#pragma unroll
      for (int i = 0; i < 8; ++i) { bih[nt][kt][i] = lo[i]; bih[nt][kt][i + 8] = hh[i]; }
    }
#pragma unroll
    for (int kt = 0; kt < 4; ++kt) {
      const _Float16* p = Whh_h + n * HDIM + kt * 32 + khalf;
      v8h lo = *(const v8h*)(p);
      v8h hh = *(const v8h*)(p + 16);
#pragma unroll
      for (int i = 0; i < 8; ++i) { bhh[nt][kt][i] = lo[i]; bhh[nt][kt][i + 8] = hh[i]; }
    }
  }

  // cell state c in registers: creg[nt][p] for unit (m=mrow+p, j=wave*8+nt*4+jj)
  float creg[2][2] = {{0.f, 0.f}, {0.f, 0.f}};

  // x element indices owned by this thread (1024 f16 per tile, 2 per thread)
  const int e0 = tid, e1 = tid + 512;
  const int m0 = e0 >> 6, f0 = e0 & 63;
  const int m1 = e1 >> 6, f1 = e1 & 63;

  // ---- stage x_0 into buffer 0 ----
  x_lds[0][e0] = (_Float16)x[((size_t)(b0 + m0) * TLEN + 0) * FDIM + f0];
  x_lds[0][e1] = (_Float16)x[((size_t)(b0 + m1) * TLEN + 0) * FDIM + f1];
  __syncthreads();

  for (int t = 0; t < TLEN; ++t) {
    const _Float16* xb = x_lds[t & 1];
    const _Float16* hb = h_lds[t & 1];
    _Float16* xn = x_lds[(t + 1) & 1];
    _Float16* hn = h_lds[(t + 1) & 1];

    // ---- prefetch x_{t+1} into registers (overlaps WMMA below) ----
    float xr0 = 0.f, xr1 = 0.f;
    if (t + 1 < TLEN) {
      xr0 = x[((size_t)(b0 + m0) * TLEN + (t + 1)) * FDIM + f0];
      xr1 = x[((size_t)(b0 + m1) * TLEN + (t + 1)) * FDIM + f1];
    }

    // ---- gates = bias + x_t @ Wih^T + h @ Whh^T (two partial accumulators) ----
    v8f accA[2], accB[2];
#pragma unroll
    for (int nt = 0; nt < 2; ++nt)
#pragma unroll
      for (int i = 0; i < 8; ++i) { accA[nt][i] = bias_n[nt]; accB[nt][i] = 0.f; }

    {
      v16h a0 = load_a_frag(xb, FDIM, lm, 0 * 32 + khalf);
      v16h a1 = load_a_frag(xb, FDIM, lm, 1 * 32 + khalf);
#pragma unroll
      for (int nt = 0; nt < 2; ++nt) {
        accA[nt] = __builtin_amdgcn_wmma_f32_16x16x32_f16(
            false, a0, false, bih[nt][0], (short)0, accA[nt], false, false);
        accB[nt] = __builtin_amdgcn_wmma_f32_16x16x32_f16(
            false, a1, false, bih[nt][1], (short)0, accB[nt], false, false);
      }
    }
#pragma unroll
    for (int kp = 0; kp < 2; ++kp) {
      v16h a0 = load_a_frag(hb, HSTR, lm, kp * 32 + khalf);
      v16h a1 = load_a_frag(hb, HSTR, lm, (kp + 2) * 32 + khalf);
#pragma unroll
      for (int nt = 0; nt < 2; ++nt) {
        accA[nt] = __builtin_amdgcn_wmma_f32_16x16x32_f16(
            false, a0, false, bhh[nt][kp], (short)0, accA[nt], false, false);
        accB[nt] = __builtin_amdgcn_wmma_f32_16x16x32_f16(
            false, a1, false, bhh[nt][kp + 2], (short)0, accB[nt], false, false);
      }
    }

    // ---- commit prefetched x_{t+1} into the other buffer (no race) ----
    if (t + 1 < TLEN) {
      xn[e0] = (_Float16)xr0;
      xn[e1] = (_Float16)xr1;
    }

    // ---- wave-private cell update: DPP quad transpose + LSTM elementwise ----
#pragma unroll
    for (int nt = 0; nt < 2; ++nt) {
      const int jg = wave * 8 + nt * 4 + jj;   // this lane's hidden unit
      // D[i][p]: cell i = row-pair element; this lane holds gate 'gl'
      float D[4][2], E[4][2], F[4][2];
#pragma unroll
      for (int i = 0; i < 4; ++i) {
        D[i][0] = accA[nt][2 * i]     + accB[nt][2 * i];
        D[i][1] = accA[nt][2 * i + 1] + accB[nt][2 * i + 1];
      }
      // 4x4 quad transpose (gates x row-pairs): DPP butterfly, masks 1 then 2
#pragma unroll
      for (int i = 0; i < 4; ++i)
#pragma unroll
        for (int p = 0; p < 2; ++p) {
          float o1 = qperm<0xB1>(D[i ^ 1][p]);         // xor 1: [1,0,3,2]
          E[i][p] = ((gl ^ i) & 1) ? o1 : D[i][p];
        }
#pragma unroll
      for (int i = 0; i < 4; ++i)
#pragma unroll
        for (int p = 0; p < 2; ++p) {
          float o2 = qperm<0x4E>(E[i ^ 2][p]);         // xor 2: [2,3,0,1]
          F[i][p] = ((gl ^ i) & 2) ? o2 : E[i][p];
        }
      // F[q][p] = gate q (0=i,1=f,2=g,3=o) for row m = mrow + p
#pragma unroll
      for (int p = 0; p < 2; ++p) {
        const int m = mrow + p;
        const float i_ = fast_sigmoid(F[0][p]);
        const float f_ = fast_sigmoid(F[1][p]);
        const float g_ = fast_tanh(F[2][p]);
        const float o_ = fast_sigmoid(F[3][p]);
        const float c  = fmaf(f_, creg[nt][p], i_ * g_);
        const float h  = o_ * fast_tanh(c);
        creg[nt][p] = c;
        hn[m * HSTR + jg] = (_Float16)h;
        if (t == TLEN - 1) h_last[(size_t)(b0 + m) * HDIM + jg] = h;
      }
    }
    __syncthreads();
  }
}

// ---------------------------------------------------------------------------
// Head: Linear(128,32) -> ReLU -> Linear(32,1) -> Sigmoid. Negligible work.
// ---------------------------------------------------------------------------
__global__ void lstm_head_kernel(const float* __restrict__ h_last,
                                 const float* __restrict__ W1,
                                 const float* __restrict__ b1,
                                 const float* __restrict__ W2,
                                 const float* __restrict__ b2,
                                 float* __restrict__ out) {
  int b = blockIdx.x * blockDim.x + threadIdx.x;
  if (b >= BSZ) return;
  const float* hr = h_last + (size_t)b * HDIM;
  float acc2 = b2[0];
#pragma unroll 4
  for (int j = 0; j < 32; ++j) {
    float a = b1[j];
    const float* w = W1 + j * HDIM;
    for (int k = 0; k < HDIM; ++k) a = fmaf(hr[k], w[k], a);
    acc2 = fmaf(fmaxf(a, 0.f), W2[j], acc2);
  }
  out[b] = fast_sigmoid(acc2);
}

// ---------------------------------------------------------------------------
// Launch
// ---------------------------------------------------------------------------
extern "C" void kernel_launch(void* const* d_in, const int* in_sizes, int n_in,
                              void* d_out, int out_size, void* d_ws, size_t ws_size,
                              hipStream_t stream) {
  const float* x    = (const float*)d_in[0];
  const float* W_ih = (const float*)d_in[1];
  const float* W_hh = (const float*)d_in[2];
  const float* b_ih = (const float*)d_in[3];
  const float* b_hh = (const float*)d_in[4];
  const float* W1   = (const float*)d_in[5];
  const float* b1   = (const float*)d_in[6];
  const float* W2   = (const float*)d_in[7];
  const float* b2   = (const float*)d_in[8];
  float* out = (float*)d_out;

  // workspace layout
  char* ws = (char*)d_ws;
  _Float16* Wih_h = (_Float16*)(ws);                         //  65536 B
  _Float16* Whh_h = (_Float16*)(ws + 65536);                 // 131072 B
  float*    bias  = (float*)   (ws + 65536 + 131072);        //   2048 B
  float*    hlast = (float*)   (ws + 65536 + 131072 + 2048); // 1 MiB

  const int prep_n = G4H * FDIM + G4H * HDIM + G4H;          // 98816
  lstm_prep_kernel<<<(prep_n + 255) / 256, 256, 0, stream>>>(
      W_ih, W_hh, b_ih, b_hh, Wih_h, Whh_h, bias);

  lstm_main_kernel<<<BSZ / 16, 512, 0, stream>>>(x, Wih_h, Whh_h, bias, hlast);

  lstm_head_kernel<<<(BSZ + 255) / 256, 256, 0, stream>>>(
      hlast, W1, b1, W2, b2, out);
}
